// DeepSeekHybridEPModule_6975026888720
// MI455X (gfx1250) — matrix operand order repaired and checked by reading, not compiled, not run on hardware
//
#include <hip/hip_runtime.h>
#include <math.h>

// ---------------- problem constants ----------------
#define TN 4096   // tokens
#define TH 512    // hidden
#define TF 2048   // ffn
#define TE 8      // experts
#define TK 2      // top-k
#define MAX_TILES 520          // max sum_e ceil(cnt_e/16)  (<= 4096*2/16 + 8)
#define MAX_SLOTS (MAX_TILES*16)
#define APAD 8                 // LDS row pad (bf16 elems) -> 4-bank stagger, no conflicts

// ---------------- types ----------------
typedef __attribute__((ext_vector_type(16))) __bf16 v16bf;
typedef __attribute__((ext_vector_type(8)))  float  v8f;
typedef __attribute__((ext_vector_type(8)))  unsigned short u16x8;

union FragAB { v16bf v; u16x8 h[2]; };

#define DEVINL __device__ __forceinline__

DEVINL unsigned short f32_to_bf16_rne(float f) {
  union { float f; unsigned u; } x; x.f = f;
  unsigned u = x.u;
  if ((u & 0x7fffffffu) > 0x7f800000u) return (unsigned short)((u >> 16) | 0x40); // quiet NaN
  u += 0x7fffu + ((u >> 16) & 1u);
  return (unsigned short)(u >> 16);
}
DEVINL float bf16_to_f32(unsigned short h) {
  union { unsigned u; float f; } x; x.u = ((unsigned)h) << 16;
  return x.f;
}

// ---- gfx1250 async copy: global -> LDS, tracked by ASYNCcnt (ISA 15.18.3 op 98) ----
DEVINL void async_g2l_b128(const unsigned short* lds_dst, const unsigned short* gsrc) {
  unsigned int  loff = (unsigned int)(unsigned long long)lds_dst;   // addr[31:0] = LDS offset
  unsigned long long ga = (unsigned long long)gsrc;
  asm volatile("global_load_async_to_lds_b128 %0, %1, off"
               :: "v"(loff), "v"(ga) : "memory");
}
DEVINL void async_wait0() {
  asm volatile("s_wait_asynccnt 0" ::: "memory");
}

// ---------------- elementwise convert ----------------
__global__ void k_f32_to_bf16(const float* __restrict__ src,
                              unsigned short* __restrict__ dst, int n) {
  int i = blockIdx.x * blockDim.x + threadIdx.x;
  if (i < n) dst[i] = f32_to_bf16_rne(src[i]);
}

// ---- fragment helpers: wave32 layouts (C per ISA 7.12.2; A/B use this code's contiguous K, not the ISA order) ----
// A (16x32 bf16): lane m = lane&15; 16 contiguous K, half selected by lane/16
// B (32x16 bf16): lane n = lane&15; 16 contiguous K, half selected by lane/16
// C (16x16 f32): vgpr r -> M = r + 8*(lane/16), N = lane&15
DEVINL v16bf load_fragA_lds(const unsigned short* arow, int k, int koffA) {
  FragAB fa;
  fa.h[0] = *(const u16x8*)(arow + k + koffA);
  fa.h[1] = *(const u16x8*)(arow + k + koffA + 8);
  return fa.v;
}
DEVINL v16bf load_fragB(const unsigned short* brow, int k, int koffB) {
  FragAB fb;
  fb.h[0] = *(const u16x8*)(brow + k + koffB);
  fb.h[1] = *(const u16x8*)(brow + k + koffB + 8);
  return fb.v;
}

// ---------------- projection GEMM: C[M,512] = A[M,512] * B[512,512]^T ----------------
// grid.x = M/16, grid.y = 512/256, block = 256 (8 waves; each wave 2 n-tiles)
__global__ __launch_bounds__(256) void k_proj_gemm(
    const unsigned short* __restrict__ A, const unsigned short* __restrict__ B,
    float* __restrict__ Cf, unsigned short* __restrict__ Cb) {
  __shared__ __attribute__((aligned(16))) unsigned short As[16][TH + APAD];
  const int mbase = blockIdx.x << 4;
  // cooperative async copy of the 16xK A tile into LDS
  for (int cidx = threadIdx.x; cidx < 16 * (TH / 8); cidx += 256) {
    int r = cidx >> 6, c = (cidx & 63) << 3;
    async_g2l_b128(&As[r][c], A + (size_t)(mbase + r) * TH + c);
  }
  async_wait0();
  __syncthreads();

  const int lane  = threadIdx.x & 31;
  const int wave  = threadIdx.x >> 5;
  const int koffA = (lane < 16) ? 0 : 16;
  const int koffB = (lane >> 4) << 4;
  const int n0 = (blockIdx.y << 8) + (wave << 4);       // first n-tile
  const unsigned short* arow = &As[lane & 15][0];
  const unsigned short* b0 = B + (size_t)(n0 + (lane & 15)) * TH;
  const unsigned short* b1 = b0 + (size_t)128 * TH;     // second n-tile (+128 cols)

  v8f acc0 = {}, acc1 = {};
#pragma unroll 4
  for (int k = 0; k < TH; k += 32) {
    v16bf fa = load_fragA_lds(arow, k, koffA);
    acc0 = __builtin_amdgcn_wmma_f32_16x16x32_bf16(false, fa, false,
             load_fragB(b0, k, koffB), (short)0, acc0, false, false);
    acc1 = __builtin_amdgcn_wmma_f32_16x16x32_bf16(false, fa, false,
             load_fragB(b1, k, koffB), (short)0, acc1, false, false);
  }
  const int crow0 = mbase + ((lane >> 4) << 3);
  const int ccol  = n0 + (lane & 15);
#pragma unroll
  for (int r = 0; r < 8; ++r) {
    size_t i0 = (size_t)(crow0 + r) * TH + ccol;
    if (Cf) { Cf[i0] = acc0[r]; Cf[i0 + 128] = acc1[r]; }
    if (Cb) { Cb[i0] = f32_to_bf16_rne(acc0[r]); Cb[i0 + 128] = f32_to_bf16_rne(acc1[r]); }
  }
}

// ---------------- router: logits -> softmax -> top2 -> re-softmax ----------------
__global__ __launch_bounds__(256) void k_router(
    const unsigned short* __restrict__ h_bf, const float* __restrict__ Wr,
    int* __restrict__ top_idx, float* __restrict__ top_w, int* __restrict__ cnt) {
  __shared__ float w[TE][TH];
  for (int i = threadIdx.x; i < TE * TH; i += 256) w[i >> 9][i & (TH - 1)] = Wr[i];
  __syncthreads();
  int n = blockIdx.x * 256 + threadIdx.x;
  const unsigned short* hr = h_bf + (size_t)n * TH;
  float acc[TE];
#pragma unroll
  for (int e = 0; e < TE; ++e) acc[e] = 0.f;
  for (int k = 0; k < TH; ++k) {
    float hv = bf16_to_f32(hr[k]);
#pragma unroll
    for (int e = 0; e < TE; ++e) acc[e] += hv * w[e][k];
  }
  float m = acc[0];
#pragma unroll
  for (int e = 1; e < TE; ++e) m = fmaxf(m, acc[e]);
  float p[TE], s = 0.f;
#pragma unroll
  for (int e = 0; e < TE; ++e) { p[e] = __expf(acc[e] - m); s += p[e]; }
  float inv = 1.f / s;
#pragma unroll
  for (int e = 0; e < TE; ++e) p[e] *= inv;
  int i0 = 0;
#pragma unroll
  for (int e = 1; e < TE; ++e) if (p[e] > p[i0]) i0 = e;
  int i1 = (i0 == 0) ? 1 : 0;
#pragma unroll
  for (int e = 0; e < TE; ++e) if (e != i0 && p[e] > p[i1]) i1 = e;
  float b  = __expf(p[i1] - p[i0]);        // reference softmaxes the top-k probabilities
  float w0 = 1.f / (1.f + b), w1 = b / (1.f + b);
  top_idx[n * 2]     = i0;  top_idx[n * 2 + 1] = i1;
  top_w[n * 2]       = w0;  top_w[n * 2 + 1]   = w1;
  atomicAdd(&cnt[i0], 1);   atomicAdd(&cnt[i1], 1);
}

__global__ void k_zero8(int* cnt) { if (threadIdx.x < TE) cnt[threadIdx.x] = 0; }

// ---------------- scan: per-expert padded slot regions + tile map ----------------
__global__ void k_scan(const int* __restrict__ cnt, int* __restrict__ slot_base,
                       int* __restrict__ tile_expert, int* __restrict__ cursor,
                       int* __restrict__ slot_token, float* __restrict__ slot_weight) {
  for (int s = threadIdx.x; s < MAX_SLOTS; s += blockDim.x) {
    slot_token[s] = -1; slot_weight[s] = 0.f;
  }
  if (threadIdx.x == 0) {
    int t = 0;
    for (int e = 0; e < TE; ++e) {
      cursor[e] = 0;
      slot_base[e] = t * 16;
      int te = (cnt[e] + 15) >> 4;
      for (int i = 0; i < te; ++i) tile_expert[t + i] = e;
      t += te;
    }
    for (; t < MAX_TILES; ++t) tile_expert[t] = -1;
  }
}

__global__ __launch_bounds__(256) void k_scatter(
    const int* __restrict__ top_idx, const float* __restrict__ top_w,
    const int* __restrict__ slot_base, int* __restrict__ cursor,
    int* __restrict__ slot_token, float* __restrict__ slot_weight,
    int* __restrict__ slot_of) {
  int n = blockIdx.x * 256 + threadIdx.x;
#pragma unroll
  for (int k = 0; k < TK; ++k) {
    int e   = top_idx[n * 2 + k];
    int pos = atomicAdd(&cursor[e], 1);
    int s   = slot_base[e] + pos;
    slot_token[s]  = n;
    slot_weight[s] = top_w[n * 2 + k];
    slot_of[n * 2 + k] = s;
  }
}

// ---------------- fused gate/up + SiLU: act[slot,F] (bf16) ----------------
// grid.x = MAX_TILES, grid.y = TF/256, block = 256; wave: 2 n-tiles x {gate,up}
__global__ __launch_bounds__(256) void k_expert_gateup(
    const unsigned short* __restrict__ h_bf, const int* __restrict__ slot_token,
    const unsigned short* __restrict__ Wg, const unsigned short* __restrict__ Wu,
    const int* __restrict__ tile_expert, unsigned short* __restrict__ act) {
  const int tile = blockIdx.x;
  const int e = tile_expert[tile];
  if (e < 0) return;                       // uniform exit: EXEC all-1 for WMMA
  const int mbase = tile << 4;

  __shared__ __attribute__((aligned(16))) unsigned short As[16][TH + APAD];
  // zero-fill (pad slots must be exact zeros), then async-gather valid token rows
  {
    u16x8 z = {};
    for (int i = threadIdx.x; i < 16 * (TH + APAD) / 8; i += 256)
      ((u16x8*)&As[0][0])[i] = z;
  }
  __syncthreads();
  for (int cidx = threadIdx.x; cidx < 16 * (TH / 8); cidx += 256) {
    int r = cidx >> 6, c = (cidx & 63) << 3;
    int tok = slot_token[mbase + r];
    if (tok >= 0)
      async_g2l_b128(&As[r][c], h_bf + (size_t)tok * TH + c);
  }
  async_wait0();
  __syncthreads();

  const int lane  = threadIdx.x & 31;
  const int wave  = threadIdx.x >> 5;
  const int koffA = (lane < 16) ? 0 : 16;
  const int koffB = (lane >> 4) << 4;
  const int n0 = (blockIdx.y << 8) + (wave << 4);
  const unsigned short* arow = &As[lane & 15][0];
  const size_t ebase = (size_t)e * TF * TH;
  const unsigned short* bg0 = Wg + ebase + (size_t)(n0 + (lane & 15)) * TH;
  const unsigned short* bu0 = Wu + ebase + (size_t)(n0 + (lane & 15)) * TH;
  const unsigned short* bg1 = bg0 + (size_t)128 * TH;
  const unsigned short* bu1 = bu0 + (size_t)128 * TH;

  v8f g0 = {}, u0 = {}, g1 = {}, u1 = {};
#pragma unroll 4
  for (int k = 0; k < TH; k += 32) {
    v16bf fa = load_fragA_lds(arow, k, koffA);
    g0 = __builtin_amdgcn_wmma_f32_16x16x32_bf16(false, fa, false,
           load_fragB(bg0, k, koffB), (short)0, g0, false, false);
    u0 = __builtin_amdgcn_wmma_f32_16x16x32_bf16(false, fa, false,
           load_fragB(bu0, k, koffB), (short)0, u0, false, false);
    g1 = __builtin_amdgcn_wmma_f32_16x16x32_bf16(false, fa, false,
           load_fragB(bg1, k, koffB), (short)0, g1, false, false);
    u1 = __builtin_amdgcn_wmma_f32_16x16x32_bf16(false, fa, false,
           load_fragB(bu1, k, koffB), (short)0, u1, false, false);
  }
  const int crow0 = mbase + ((lane >> 4) << 3);
  const int ccol  = n0 + (lane & 15);
#pragma unroll
  for (int r = 0; r < 8; ++r) {
    float ga = g0[r], ua = u0[r];
    float gb = g1[r], ub = u1[r];
    float va = (ga / (1.f + __expf(-ga))) * ua;   // silu(g) * u
    float vb = (gb / (1.f + __expf(-gb))) * ub;
    size_t base = (size_t)(crow0 + r) * TF + ccol;
    act[base]       = f32_to_bf16_rne(va);
    act[base + 128] = f32_to_bf16_rne(vb);
  }
}

// ---------------- down-proj per slot tile: slot_out[slot,H] (f32) ----------------
// grid.x = MAX_TILES, grid.y = TH/256, block = 256; wave: 2 n-tiles
__global__ __launch_bounds__(256) void k_expert_down(
    const unsigned short* __restrict__ act, const unsigned short* __restrict__ Wd,
    const int* __restrict__ tile_expert, float* __restrict__ slot_out) {
  const int tile = blockIdx.x;
  const int e = tile_expert[tile];
  if (e < 0) return;
  const int mbase = tile << 4;

  __shared__ __attribute__((aligned(16))) unsigned short As[16][TF + APAD];
  for (int cidx = threadIdx.x; cidx < 16 * (TF / 8); cidx += 256) {
    int r = cidx >> 8, c = (cidx & 255) << 3;
    async_g2l_b128(&As[r][c], act + (size_t)(mbase + r) * TF + c);
  }
  async_wait0();
  __syncthreads();

  const int lane  = threadIdx.x & 31;
  const int wave  = threadIdx.x >> 5;
  const int koffA = (lane < 16) ? 0 : 16;
  const int koffB = (lane >> 4) << 4;
  const int n0 = (blockIdx.y << 8) + (wave << 4);
  const unsigned short* arow = &As[lane & 15][0];
  const unsigned short* b0 = Wd + (size_t)e * TH * TF + (size_t)(n0 + (lane & 15)) * TF;
  const unsigned short* b1 = b0 + (size_t)128 * TF;

  v8f acc0 = {}, acc1 = {};
#pragma unroll 4
  for (int k = 0; k < TF; k += 32) {
    v16bf fa = load_fragA_lds(arow, k, koffA);
    acc0 = __builtin_amdgcn_wmma_f32_16x16x32_bf16(false, fa, false,
             load_fragB(b0, k, koffB), (short)0, acc0, false, false);
    acc1 = __builtin_amdgcn_wmma_f32_16x16x32_bf16(false, fa, false,
             load_fragB(b1, k, koffB), (short)0, acc1, false, false);
  }
  const int crow0 = mbase + ((lane >> 4) << 3);
  const int ccol  = n0 + (lane & 15);
#pragma unroll
  for (int r = 0; r < 8; ++r) {
    size_t i0 = (size_t)(crow0 + r) * TH + ccol;
    slot_out[i0]       = acc0[r];
    slot_out[i0 + 128] = acc1[r];
  }
}

// ---------------- top-2 combine (no atomics; per-token slot pointers) ----------------
__global__ __launch_bounds__(256) void k_combine(
    const float* __restrict__ slot_out, const int* __restrict__ slot_of,
    const float* __restrict__ slot_weight, unsigned short* __restrict__ comb_bf) {
  int i = blockIdx.x * 256 + threadIdx.x;       // TN*TH
  int n = i >> 9, c = i & (TH - 1);
  int s0 = slot_of[n * 2], s1 = slot_of[n * 2 + 1];
  float v = slot_weight[s0] * slot_out[(size_t)s0 * TH + c] +
            slot_weight[s1] * slot_out[(size_t)s1 * TH + c];
  comb_bf[i] = f32_to_bf16_rne(v);
}

// ---------------- host ----------------
static inline size_t align256(size_t x) { return (x + 255) & ~(size_t)255; }

extern "C" void kernel_launch(void* const* d_in, const int* in_sizes, int n_in,
                              void* d_out, int out_size, void* d_ws, size_t ws_size,
                              hipStream_t stream) {
  (void)in_sizes; (void)n_in; (void)out_size; (void)ws_size;
  const float* x    = (const float*)d_in[0];
  const float* Win  = (const float*)d_in[1];
  const float* Wr   = (const float*)d_in[2];
  const float* Wg   = (const float*)d_in[3];
  const float* Wu   = (const float*)d_in[4];
  const float* Wd   = (const float*)d_in[5];
  const float* Wout = (const float*)d_in[6];
  float* out = (float*)d_out;

  char* ws = (char*)d_ws;
  size_t off = 0;
  auto alloc = [&](size_t bytes) { char* p = ws + off; off = align256(off + bytes); return p; };

  unsigned short* x_bf    = (unsigned short*)alloc((size_t)TN * TH * 2);
  unsigned short* Win_bf  = (unsigned short*)alloc((size_t)TH * TH * 2);
  unsigned short* Wg_bf   = (unsigned short*)alloc((size_t)TE * TF * TH * 2);
  unsigned short* Wu_bf   = (unsigned short*)alloc((size_t)TE * TF * TH * 2);
  unsigned short* Wd_bf   = (unsigned short*)alloc((size_t)TE * TH * TF * 2);
  unsigned short* Wout_bf = (unsigned short*)alloc((size_t)TH * TH * 2);
  unsigned short* h_bf    = (unsigned short*)alloc((size_t)TN * TH * 2);
  int*   top_idx  = (int*)  alloc((size_t)TN * 2 * 4);
  float* top_w    = (float*)alloc((size_t)TN * 2 * 4);
  int*   cnt      = (int*)  alloc(TE * 4);
  int*   cursor   = (int*)  alloc(TE * 4);
  int*   slot_base= (int*)  alloc(TE * 4);
  int*   tile_exp = (int*)  alloc(MAX_TILES * 4);
  int*   slot_tok = (int*)  alloc(MAX_SLOTS * 4);
  float* slot_wt  = (float*)alloc(MAX_SLOTS * 4);
  int*   slot_of  = (int*)  alloc((size_t)TN * 2 * 4);
  unsigned short* act_ws   = (unsigned short*)alloc((size_t)MAX_SLOTS * TF * 2);
  float* slot_out = (float*)alloc((size_t)MAX_SLOTS * TH * 4);
  unsigned short* comb_bf  = (unsigned short*)alloc((size_t)TN * TH * 2);

  const int thr = 256;
  auto cvt = [&](const float* s, unsigned short* d, size_t n) {
    k_f32_to_bf16<<<(unsigned)((n + thr - 1) / thr), thr, 0, stream>>>(s, d, (int)n);
  };
  cvt(x,    x_bf,    (size_t)TN * TH);
  cvt(Win,  Win_bf,  (size_t)TH * TH);
  cvt(Wg,   Wg_bf,   (size_t)TE * TF * TH);
  cvt(Wu,   Wu_bf,   (size_t)TE * TF * TH);
  cvt(Wd,   Wd_bf,   (size_t)TE * TH * TF);
  cvt(Wout, Wout_bf, (size_t)TH * TH);

  // h = x @ Win^T  -> bf16
  k_proj_gemm<<<dim3(TN / 16, TH / 256), 256, 0, stream>>>(x_bf, Win_bf, nullptr, h_bf);

  k_zero8<<<1, 32, 0, stream>>>(cnt);
  k_router<<<TN / 256, 256, 0, stream>>>(h_bf, Wr, top_idx, top_w, cnt);
  k_scan<<<1, 256, 0, stream>>>(cnt, slot_base, tile_exp, cursor, slot_tok, slot_wt);
  k_scatter<<<TN / 256, 256, 0, stream>>>(top_idx, top_w, slot_base, cursor,
                                          slot_tok, slot_wt, slot_of);

  k_expert_gateup<<<dim3(MAX_TILES, TF / 256), 256, 0, stream>>>(
      h_bf, slot_tok, Wg_bf, Wu_bf, tile_exp, act_ws);
  k_expert_down<<<dim3(MAX_TILES, TH / 256), 256, 0, stream>>>(
      act_ws, Wd_bf, tile_exp, slot_out);

  k_combine<<<(TN * TH) / 256, 256, 0, stream>>>(slot_out, slot_of, slot_wt, comb_bf);

  // out = combined @ Wout^T  -> f32
  k_proj_gemm<<<dim3(TN / 16, TH / 256), 256, 0, stream>>>(comb_bf, Wout_bf, out, nullptr);
}
